// TransformerLayer_64596308131864
// MI455X (gfx1250) — matrix operand order repaired and checked by reading, not compiled, not run on hardware
//
#include <hip/hip_runtime.h>
#include <hip/hip_bf16.h>
#include <math.h>

// ---------------- problem constants ----------------
#define NB   4
#define SQ   2048
#define DM   512
#define HH   8
#define DK   64
#define DFF  2048
#define MROWS (NB*SQ)          // 8192
#define QKVW  (3*DM)           // 1536

typedef __bf16 bf16;
typedef __attribute__((ext_vector_type(8)))  bf16  v8bf;
typedef __attribute__((ext_vector_type(16))) bf16  v16bf;
typedef __attribute__((ext_vector_type(8)))  float v8f;

union BF16x16 { v16bf v; v8bf h[2]; };

// Issue one 16B async global->LDS copy (tracked by ASYNCcnt).
__device__ __forceinline__ void async_copy16(const bf16* gptr, bf16* lptr) {
    unsigned lds = (unsigned)(size_t)lptr;   // flat low 32 bits == LDS byte address
    asm volatile("global_load_async_to_lds_b128 %0, %1, off"
                 :: "v"(lds), "v"(gptr) : "memory");
}

// ---------------- weight packing (all transposed: BT[N][K]) ----------------
// Wq/Wk/Wv: [H, D, DK] fp32 -> WqkvT_bf16 [1536, 512] ; row c = which*512 + h*64 + k, col d
// Q weights are pre-scaled by 1/sqrt(DK) so attention logits need no scaling.
__global__ void pack_qkv_wT_kernel(const float* __restrict__ Wq,
                                   const float* __restrict__ Wk,
                                   const float* __restrict__ Wv,
                                   bf16* __restrict__ out) {
    int idx = blockIdx.x * 256 + threadIdx.x;
    if (idx >= QKVW * DM) return;
    int c = idx >> 9, d = idx & 511;
    int which = c >> 9;          // 0=Q,1=K,2=V
    int cc = c & 511;
    int h = cc >> 6, k = cc & 63;
    const float* W = (which == 0) ? Wq : (which == 1) ? Wk : Wv;
    float scl = (which == 0) ? 0.125f : 1.0f;     // 1/sqrt(64)
    out[idx] = (bf16)(scl * W[((size_t)h * DM + d) * DK + k]);
}

// Wo: [H*DK, D] fp32 with row f = v*H + h  ->  WoT [512(c)][512(r=h*64+v)]
__global__ void pack_woT_kernel(const float* __restrict__ Wo, bf16* __restrict__ out) {
    int idx = blockIdx.x * 256 + threadIdx.x;
    if (idx >= DM * DM) return;
    int c = idx >> 9, r = idx & 511;
    int h = r >> 6, v = r & 63;
    out[idx] = (bf16)Wo[(size_t)(v * HH + h) * DM + c];
}

// generic fp32 [K][N] -> bf16 transposed [N][K]
__global__ void cvtT_f32_bf16_kernel(const float* __restrict__ in, bf16* __restrict__ out,
                                     int K, int N) {
    int idx = blockIdx.x * 256 + threadIdx.x;
    if (idx >= K * N) return;
    int n = idx / K, k = idx % K;
    out[idx] = (bf16)in[(size_t)k * N + n];
}

// qkv V-region [N*S,1536] -> Vt [NB*HH][DK][SQ] via LDS tile transpose
__global__ __launch_bounds__(256) void v_transpose_kernel(const bf16* __restrict__ qkv,
                                                          bf16* __restrict__ vt) {
    __shared__ __align__(16) bf16 T[64 * 40];           // [d][tok]
    const int nh = blockIdx.y;                          // n*8+h
    const int n = nh >> 3, h = nh & 7;
    const int s0 = blockIdx.x * 32;
    const int t = threadIdx.x;
    {   // read 32 tokens x 64 d, coalesced along d
        int tok = t >> 3, dch = (t & 7) * 8;
        v8bf v = *(const v8bf*)&qkv[((size_t)n * SQ + s0 + tok) * QKVW + 2 * DM + h * DK + dch];
#pragma unroll
        for (int e = 0; e < 8; ++e) T[(dch + e) * 40 + tok] = v[e];
    }
    __syncthreads();
    {   // write 64 d-rows x 32 tokens, coalesced along s
        int d = t >> 2, tch = (t & 3) * 8;
        v8bf v = *(const v8bf*)&T[d * 40 + tch];
        *(v8bf*)&vt[((size_t)nh * DK + d) * SQ + s0 + tch] = v;
    }
}

// ---------------- LayerNorm (ddof=1, eps added to std), fp32 in -> bf16 out ----------------
__global__ __launch_bounds__(256) void ln_kernel(const float* __restrict__ x,
                                                 const float* __restrict__ g,
                                                 const float* __restrict__ b,
                                                 bf16* __restrict__ out) {
    const int row = blockIdx.x;
    const float* xr = x + (size_t)row * DM;
    const int t = threadIdx.x;
    float a0 = xr[t], a1 = xr[t + 256];

    __shared__ float red[8];
    float s = a0 + a1;
    for (int off = 16; off; off >>= 1) s += __shfl_xor(s, off, 32);
    int w = t >> 5, l = t & 31;
    if (l == 0) red[w] = s;
    __syncthreads();
    float tot = 0.f;
    for (int i = 0; i < 8; ++i) tot += red[i];
    float mean = tot * (1.0f / DM);
    float d0 = a0 - mean, d1 = a1 - mean;
    float sq = d0 * d0 + d1 * d1;
    for (int off = 16; off; off >>= 1) sq += __shfl_xor(sq, off, 32);
    __syncthreads();
    if (l == 0) red[w] = sq;
    __syncthreads();
    float tots = 0.f;
    for (int i = 0; i < 8; ++i) tots += red[i];
    float inv = 1.0f / (sqrtf(tots / (float)(DM - 1)) + 1e-10f);
    out[(size_t)row * DM + t]       = (bf16)(d0 * inv * g[t] + b[t]);
    out[(size_t)row * DM + t + 256] = (bf16)(d1 * inv * g[t + 256] + b[t + 256]);
}

// ---------------- bf16 WMMA GEMM: C = A[M,K] * BT[N,K]^T, 128x128 tile, BK=32 ----------
// Double-buffered LDS, tiles staged via global_load_async_to_lds_b128.
// MODE 0: C_bf16 = A*B ; 1: C_f32 = Res + A*B ; 2: C_bf16 = gelu(A*B+Bias) ; 3: C_f32 = Res+A*B+Bias
template <int MODE>
__global__ __launch_bounds__(256) void gemm_bf16_kernel(const bf16* __restrict__ A,
                                                        const bf16* __restrict__ BT,
                                                        void* __restrict__ Cout,
                                                        const float* __restrict__ Res,
                                                        const float* __restrict__ Bias,
                                                        int M, int N, int K) {
    __shared__ __align__(16) bf16 Alds[2][128 * 40];
    __shared__ __align__(16) bf16 Blds[2][128 * 40];   // [n][k]

    const int tid  = threadIdx.x;
    const int lane = tid & 31, w = tid >> 5;
    const int lrow = lane & 15, lhi = lane >> 4;
    const int m0 = blockIdx.y * 128, n0 = blockIdx.x * 128;
    const int mw = (w & 3) * 32, nw = (w >> 2) * 64;

    // this thread's two 16B staging chunks: chunk c -> row r = c>>2, col off = (c&3)*8
    const int r0 = tid >> 2,         co0 = (tid & 3) * 8;
    const int r1 = (tid + 256) >> 2, co1 = ((tid + 256) & 3) * 8;

    auto stage = [&](int buf, int k0) {
        async_copy16(&A [(size_t)(m0 + r0) * K + k0 + co0], &Alds[buf][r0 * 40 + co0]);
        async_copy16(&A [(size_t)(m0 + r1) * K + k0 + co1], &Alds[buf][r1 * 40 + co1]);
        async_copy16(&BT[(size_t)(n0 + r0) * K + k0 + co0], &Blds[buf][r0 * 40 + co0]);
        async_copy16(&BT[(size_t)(n0 + r1) * K + k0 + co1], &Blds[buf][r1 * 40 + co1]);
    };

    v8f acc[2][4];
#pragma unroll
    for (int mt = 0; mt < 2; ++mt)
#pragma unroll
        for (int nt = 0; nt < 4; ++nt)
#pragma unroll
            for (int i = 0; i < 8; ++i) acc[mt][nt][i] = 0.0f;

    stage(0, 0);
    const int nsteps = K >> 5;
    for (int it = 0; it < nsteps; ++it) {
        const int buf = it & 1;
        if (it + 1 < nsteps) {
            stage(buf ^ 1, (it + 1) << 5);
            asm volatile("s_wait_asynccnt 0x4" ::: "memory");  // previous stage complete
        } else {
            asm volatile("s_wait_asynccnt 0x0" ::: "memory");
        }
        __syncthreads();   // staged tile visible to all waves

        BF16x16 afr[2], bfr[4];
#pragma unroll
        for (int mt = 0; mt < 2; ++mt) {
            const bf16* p = &Alds[buf][(mw + mt * 16 + lrow) * 40];
            afr[mt].h[0] = *(const v8bf*)(p + lhi * 16);      // K = {0..7 | 16..23}
            afr[mt].h[1] = *(const v8bf*)(p + lhi * 16 + 8);  // K = {8..15 | 24..31}
        }
#pragma unroll
        for (int nt = 0; nt < 4; ++nt) {
            const bf16* p = &Blds[buf][(nw + nt * 16 + lrow) * 40] + lhi * 16;
            bfr[nt].h[0] = *(const v8bf*)p;
            bfr[nt].h[1] = *(const v8bf*)(p + 8);
        }
#pragma unroll
        for (int mt = 0; mt < 2; ++mt)
#pragma unroll
            for (int nt = 0; nt < 4; ++nt)
                acc[mt][nt] = __builtin_amdgcn_wmma_f32_16x16x32_bf16(
                    false, afr[mt].v, false, bfr[nt].v, (short)0, acc[mt][nt], false, false);

        __syncthreads();   // all waves done reading buf before it is restaged
    }

    // epilogue
#pragma unroll
    for (int mt = 0; mt < 2; ++mt)
#pragma unroll
        for (int nt = 0; nt < 4; ++nt)
#pragma unroll
            for (int i = 0; i < 8; ++i) {
                int rg = m0 + mw + mt * 16 + lhi * 8 + i;
                int cg = n0 + nw + nt * 16 + lrow;
                size_t o = (size_t)rg * N + cg;
                float v = acc[mt][nt][i];
                if (MODE == 0) {
                    ((bf16*)Cout)[o] = (bf16)v;
                } else if (MODE == 1) {
                    ((float*)Cout)[o] = Res[o] + v;
                } else if (MODE == 2) {
                    float xx = v + Bias[cg];
                    ((bf16*)Cout)[o] = (bf16)(0.5f * xx * (1.0f + erff(xx * 0.70710678118654752f)));
                } else {
                    ((float*)Cout)[o] = Res[o] + v + Bias[cg];
                }
            }
}

// ---------------- flash attention (causal), barrier-free per-wave ----------------
// qkv: [N*S, 1536] (Q pre-scaled at h*64+d, K at 512+h*64+d) ; vt: [N*H][DK][SQ]
// heads: [N*S, 512] col = h*64+d
__global__ __launch_bounds__(256) void attn_kernel(const bf16* __restrict__ qkv,
                                                   const bf16* __restrict__ vt,
                                                   bf16* __restrict__ heads) {
    __shared__ __align__(16) bf16 Plds[8][16 * 40];   // per-wave P tile [16 q][32 key]

    const int tid = threadIdx.x, lane = tid & 31, w = tid >> 5;
    const int lrow = lane & 15, lhi = lane >> 4;
    const int nh = blockIdx.y;
    const int n = nh >> 3, h = nh & 7;
    const int q0 = blockIdx.x * 128 + w * 16;          // this wave's first query row

    const bf16* Qb = qkv + (size_t)n * SQ * QKVW + h * DK;
    const bf16* Kb = Qb + DM;
    const bf16* Vb = vt + (size_t)nh * DK * SQ;

    // Q fragments (A-layout), two K-steps over d=0..63
    BF16x16 qfr[2];
#pragma unroll
    for (int s = 0; s < 2; ++s) {
        const bf16* p = Qb + (size_t)(q0 + lrow) * QKVW + s * 32;
        qfr[s].h[0] = *(const v8bf*)(p + lhi * 16);
        qfr[s].h[1] = *(const v8bf*)(p + lhi * 16 + 8);
    }

    float mi[8], li[8];                                // li = per-LANE partial row sum
    v8f Oacc[4];
#pragma unroll
    for (int i = 0; i < 8; ++i) { mi[i] = -1e30f; li[i] = 0.0f; }
#pragma unroll
    for (int j = 0; j < 4; ++j)
#pragma unroll
        for (int i = 0; i < 8; ++i) Oacc[j][i] = 0.0f;

    auto step = [&](int t0, bool masked) {
        // scores S = Q K^T for two 16-wide key tiles (K frags straight from global)
        v8f sc[2];
#pragma unroll
        for (int nt = 0; nt < 2; ++nt) {
#pragma unroll
            for (int i = 0; i < 8; ++i) sc[nt][i] = 0.0f;
#pragma unroll
            for (int s = 0; s < 2; ++s) {
                BF16x16 kf;
                const bf16* kp = Kb + (size_t)(t0 + nt * 16 + lrow) * QKVW + s * 32 + lhi * 16;
                kf.h[0] = *(const v8bf*)kp;
                kf.h[1] = *(const v8bf*)(kp + 8);
                sc[nt] = __builtin_amdgcn_wmma_f32_16x16x32_bf16(
                    false, qfr[s].v, false, kf.v, (short)0, sc[nt], false, false);
            }
        }
        // causal mask only on the diagonal block
        if (masked) {
            const float NEGINF = -__builtin_inff();
#pragma unroll
            for (int i = 0; i < 8; ++i) {
                int r = q0 + lhi * 8 + i;
                if (t0 + lrow > r)      sc[0][i] = NEGINF;
                if (t0 + 16 + lrow > r) sc[1][i] = NEGINF;
            }
        }
        // online softmax: max reduced across 16 lanes; sum kept as per-lane partial
        float scl[8];
#pragma unroll
        for (int i = 0; i < 8; ++i) {
            float m = fmaxf(sc[0][i], sc[1][i]);
            for (int off = 8; off; off >>= 1) m = fmaxf(m, __shfl_xor(m, off, 32));
            float mnew = fmaxf(mi[i], m);
            float rescale = __expf(mi[i] - mnew);
            mi[i] = mnew;
            float p0 = __expf(sc[0][i] - mnew);
            float p1 = __expf(sc[1][i] - mnew);
            sc[0][i] = p0; sc[1][i] = p1;
            li[i] = li[i] * rescale + p0 + p1;
            scl[i] = rescale;
        }
#pragma unroll
        for (int j = 0; j < 4; ++j)
#pragma unroll
            for (int i = 0; i < 8; ++i) Oacc[j][i] *= scl[i];

        // C-layout -> A-layout for P via per-wave LDS round trip
        bf16* P = &Plds[w][0];
#pragma unroll
        for (int i = 0; i < 8; ++i) {
            P[(lhi * 8 + i) * 40 + lrow]      = (bf16)sc[0][i];
            P[(lhi * 8 + i) * 40 + 16 + lrow] = (bf16)sc[1][i];
        }
        asm volatile("s_wait_dscnt 0x0" ::: "memory");
        BF16x16 pf;
        pf.h[0] = *(const v8bf*)&P[lrow * 40 + lhi * 16];
        pf.h[1] = *(const v8bf*)&P[lrow * 40 + lhi * 16 + 8];

        // O += P @ V  (V^T frags straight from vt: contiguous tokens per lane)
#pragma unroll
        for (int j = 0; j < 4; ++j) {
            BF16x16 vf;
            const bf16* vp = Vb + (size_t)(j * 16 + lrow) * SQ + t0 + lhi * 16;
            vf.h[0] = *(const v8bf*)vp;
            vf.h[1] = *(const v8bf*)(vp + 8);
            Oacc[j] = __builtin_amdgcn_wmma_f32_16x16x32_bf16(
                false, pf.v, false, vf.v, (short)0, Oacc[j], false, false);
        }
    };

    const int nfull = q0 >> 5;                 // fully-unmasked 32-key blocks
    for (int kb = 0; kb < nfull; ++kb) step(kb * 32, false);
    step(nfull * 32, true);                    // single diagonal (masked) block

    // final: reduce per-lane partial sums across the row, normalize, store
    float inv[8];
#pragma unroll
    for (int i = 0; i < 8; ++i) {
        float rs = li[i];
        for (int off = 8; off; off >>= 1) rs += __shfl_xor(rs, off, 32);
        inv[i] = 1.0f / rs;
    }
#pragma unroll
    for (int j = 0; j < 4; ++j)
#pragma unroll
        for (int i = 0; i < 8; ++i) {
            int rg = q0 + lhi * 8 + i;
            heads[((size_t)n * SQ + rg) * DM + h * DK + j * 16 + lrow] =
                (bf16)(Oacc[j][i] * inv[i]);
        }
}

// ---------------- host-side launcher ----------------
extern "C" void kernel_launch(void* const* d_in, const int* in_sizes, int n_in,
                              void* d_out, int out_size, void* d_ws, size_t ws_size,
                              hipStream_t stream) {
    const float* X    = (const float*)d_in[0];
    const float* Wq   = (const float*)d_in[1];
    const float* Wk   = (const float*)d_in[2];
    const float* Wv   = (const float*)d_in[3];
    const float* Wo   = (const float*)d_in[4];
    const float* g1   = (const float*)d_in[5];
    const float* b1   = (const float*)d_in[6];
    const float* g2   = (const float*)d_in[7];
    const float* b2   = (const float*)d_in[8];
    const float* W1   = (const float*)d_in[9];
    const float* bff1 = (const float*)d_in[10];
    const float* W2   = (const float*)d_in[11];
    const float* bff2 = (const float*)d_in[12];

    // workspace carve (256B aligned)
    char* ws = (char*)d_ws;
    size_t off = 0;
    auto carve = [&](size_t bytes) { void* p = ws + off; off = (off + bytes + 255) & ~(size_t)255; return p; };
    bf16* y_bf     = (bf16*)carve((size_t)MROWS * DM * 2);
    bf16* WqkvT_bf = (bf16*)carve((size_t)QKVW * DM * 2);
    bf16* qkv_bf   = (bf16*)carve((size_t)MROWS * QKVW * 2);
    bf16* vt_bf    = (bf16*)carve((size_t)NB * HH * DK * SQ * 2);
    bf16* WoT_bf   = (bf16*)carve((size_t)DM * DM * 2);
    bf16* heads    = (bf16*)carve((size_t)MROWS * DM * 2);
    float* first   = (float*)carve((size_t)MROWS * DM * 4);
    bf16* t_bf     = (bf16*)carve((size_t)MROWS * DM * 2);
    bf16* W1T_bf   = (bf16*)carve((size_t)DM * DFF * 2);
    bf16* u_bf     = (bf16*)carve((size_t)MROWS * DFF * 2);
    bf16* W2T_bf   = (bf16*)carve((size_t)DFF * DM * 2);
    (void)ws_size; (void)in_sizes; (void)n_in; (void)out_size;

    // weight packing (transposed) / conversion
    pack_qkv_wT_kernel<<<(QKVW * DM + 255) / 256, 256, 0, stream>>>(Wq, Wk, Wv, WqkvT_bf);
    pack_woT_kernel<<<(DM * DM + 255) / 256, 256, 0, stream>>>(Wo, WoT_bf);
    cvtT_f32_bf16_kernel<<<(DM * DFF + 255) / 256, 256, 0, stream>>>(W1, W1T_bf, DM, DFF);
    cvtT_f32_bf16_kernel<<<(DFF * DM + 255) / 256, 256, 0, stream>>>(W2, W2T_bf, DFF, DM);

    // pre-LN
    ln_kernel<<<MROWS, 256, 0, stream>>>(X, g1, b1, y_bf);
    // QKV projection: [8192x512] x [512x1536]
    gemm_bf16_kernel<0><<<dim3(QKVW / 128, MROWS / 128), 256, 0, stream>>>(
        y_bf, WqkvT_bf, qkv_bf, nullptr, nullptr, MROWS, QKVW, DM);
    // V transpose for attention B-fragments
    v_transpose_kernel<<<dim3(SQ / 32, NB * HH), 256, 0, stream>>>(qkv_bf, vt_bf);
    // causal flash attention
    attn_kernel<<<dim3(SQ / 128, NB * HH), 256, 0, stream>>>(qkv_bf, vt_bf, heads);
    // out-proj + residual: first = X + heads @ Wo
    gemm_bf16_kernel<1><<<dim3(DM / 128, MROWS / 128), 256, 0, stream>>>(
        heads, WoT_bf, first, X, nullptr, MROWS, DM, DM);
    // second LN
    ln_kernel<<<MROWS, 256, 0, stream>>>(first, g2, b2, t_bf);
    // FFN1 + bias + exact GELU
    gemm_bf16_kernel<2><<<dim3(DFF / 128, MROWS / 128), 256, 0, stream>>>(
        t_bf, W1T_bf, u_bf, nullptr, bff1, MROWS, DFF, DM);
    // FFN2 + bias + residual -> out
    gemm_bf16_kernel<3><<<dim3(DM / 128, MROWS / 128), 256, 0, stream>>>(
        u_bf, W2T_bf, (float*)d_out, first, bff2, MROWS, DM, DFF);
}
